// SparseLocalFrameLayer_6992206758100
// MI455X (gfx1250) — compile-verified, matrix operand order checked by reading, not run on hardware
//
#include <hip/hip_runtime.h>
#include <hip/hip_bf16.h>

#define SDIM 64
#define VDIM 64
#define NN   100000
#define NE   1600000
#define KPAD 224      // 195 padded to 7 * 32
#define HID  128
#define EPB  64       // edges per block (4 M-tiles of 16)
#define ASTRIDE 232   // LDS row stride (halves); 464B = 29*16 -> aligned, conflict-free
#define HSTRIDE 136   // LDS hidden row stride (halves); 272B = 17*16

typedef __attribute__((ext_vector_type(16))) _Float16 v16h;
typedef __attribute__((ext_vector_type(8)))  _Float16 v8h;
typedef __attribute__((ext_vector_type(8)))  float    v8f;

union V16 { v16h v; v8h h[2]; };

// ---- Pack W1 (195x128 f32, row-major) into WMMA B-fragment order, f16, K padded to 224.
// Layout: [ktile(7)][ntile(8)][lane(32)][16 halves]; lane's K set follows the ISA
// 16-bit operand layout (lanes 0-15: K {0..7,16..23}; lanes 16-31: K {8..15,24..31}).
__global__ __launch_bounds__(256) void pack_w1(const float* __restrict__ W,
                                               _Float16* __restrict__ out) {
    int idx = blockIdx.x * 256 + threadIdx.x;
    if (idx >= 7 * 8 * 32 * 16) return;
    int v    = idx & 15;
    int lane = (idx >> 4) & 31;
    int nt   = (idx >> 9) & 7;
    int kt   = idx >> 12;
    int v8   = (v < 8) ? v : v + 8;
    int kloc = v8 + ((lane & 16) ? 8 : 0);
    int k    = kt * 32 + kloc;
    int n    = nt * 16 + (lane & 15);
    float val = (k < 195) ? W[k * HID + n] : 0.0f;
    out[idx] = (_Float16)val;
}

// ---- Pack W2 (128x128 f32) the same way: [ktile(4)][ntile(8)][lane(32)][16].
__global__ __launch_bounds__(256) void pack_w2(const float* __restrict__ W,
                                               _Float16* __restrict__ out) {
    int idx = blockIdx.x * 256 + threadIdx.x;
    if (idx >= 4 * 8 * 32 * 16) return;
    int v    = idx & 15;
    int lane = (idx >> 4) & 31;
    int nt   = (idx >> 9) & 7;
    int kt   = idx >> 12;
    int v8   = (v < 8) ? v : v + 8;
    int kloc = v8 + ((lane & 16) ? 8 : 0);
    int k    = kt * 32 + kloc;
    int n    = nt * 16 + (lane & 15);
    out[idx] = (_Float16)W[k * HID + n];
}

// ---- out = concat(h_s, h_v) (residual term; atomics accumulate on top).
__global__ __launch_bounds__(256) void init_out(const float* __restrict__ hs,
                                                const float* __restrict__ hv,
                                                float* __restrict__ out) {
    int i = blockIdx.x * 256 + threadIdx.x;
    const int total = NN * SDIM;
    if (i < total) {
        out[i]         = hs[i];
        out[total + i] = hv[i];
    }
}

// ---- Main edge kernel: 64 edges per block, 8 waves, wave w owns N-tile w.
__global__ __launch_bounds__(256)
void edge_kernel(const float* __restrict__ hs, const float* __restrict__ hv,
                 const float* __restrict__ pos, const float* __restrict__ ori,
                 const float* __restrict__ b1,  const float* __restrict__ b2,
                 const int*   __restrict__ eidx,
                 const _Float16* __restrict__ W1p, const _Float16* __restrict__ W2p,
                 float* __restrict__ out) {
    __shared__ __align__(16) _Float16 As[EPB][ASTRIDE];     // 16-edge x 224-K A tiles (f16)
    __shared__ __align__(16) _Float16 Hs[4][16][HSTRIDE];   // per-M-tile hidden (f16)
    __shared__ int   sSrc[EPB], sDst[EPB];
    __shared__ float sC[EPB], sS[EPB], sDist[EPB], sCg[EPB], sSg[EPB];

    const int tid   = threadIdx.x;
    const int ebase = blockIdx.x * EPB;

    // Per-edge geometry scalars.
    if (tid < EPB) {
        int E   = ebase + tid;
        int src = eidx[E];
        int dst = eidx[NE + E];
        sSrc[tid] = src; sDst[tid] = dst;
        float dx = pos[2 * src]     - pos[2 * dst];
        float dy = pos[2 * src + 1] - pos[2 * dst + 1];
        sDist[tid] = sqrtf(dx * dx + dy * dy) + 1e-6f;
        float phi   = atan2f(dy, dx);
        float alpha = ori[dst];
        float beta  = ori[src];
        float dphi2 = 2.0f * (phi - alpha);   // SPIN = 2
        float rot   = 2.0f * (beta - alpha);
        sCg[tid] = cosf(dphi2); sSg[tid] = sinf(dphi2);
        sC[tid]  = cosf(rot);   sS[tid]  = sinf(rot);
    }
    __syncthreads();

    // Build A: msg_input = [h_s[src](64) | h_s[dst](64) | v_rot(64) | geo(3) | pad(29)]
    for (int idx = tid; idx < EPB * KPAD; idx += 256) {
        int e = idx / KPAD;
        int c = idx - e * KPAD;
        int src = sSrc[e], dst = sDst[e];
        float val;
        if (c < 64) {
            val = hs[src * SDIM + c];
        } else if (c < 128) {
            val = hs[dst * SDIM + (c - 64)];
        } else if (c < 192) {
            int i = c - 128;
            int p = i >> 1;
            float vx = hv[src * VDIM + 2 * p];
            float vy = hv[src * VDIM + 2 * p + 1];
            val = (i & 1) ? (vx * sS[e] + vy * sC[e])
                          : (vx * sC[e] - vy * sS[e]);
        } else if (c == 192) val = sDist[e];
        else if   (c == 193) val = sCg[e];
        else if   (c == 194) val = sSg[e];
        else                 val = 0.0f;
        As[e][c] = (_Float16)val;
    }
    __syncthreads();

    const int w    = tid >> 5;   // wave id == N-tile
    const int l    = tid & 31;   // lane
    const int hi   = l >> 4;     // half-wave select (K split for A, M split for C/D)
    const int nl   = l & 15;     // column within N-tile / A-row (M)
    const int mrow = hi * 8;     // first C/D row this lane holds

    // B fragments live in VGPRs for the whole block.
    V16 B1f[7], B2f[4];
#pragma unroll
    for (int k = 0; k < 7; ++k)
        B1f[k].v = *(const v16h*)(W1p + (((k * 8 + w) * 32 + l) << 4));
#pragma unroll
    for (int k = 0; k < 4; ++k)
        B2f[k].v = *(const v16h*)(W2p + (((k * 8 + w) * 32 + l) << 4));

    const float bias1 = b1[w * 16 + nl];
    const float bias2 = b2[w * 16 + nl];

    // GEMM1 (16x224 @ 224x16 per wave) + SiLU -> LDS hidden.
#pragma unroll
    for (int t = 0; t < 4; ++t) {
        v8f acc = {};
#pragma unroll
        for (int k = 0; k < 7; ++k) {
            V16 a;
            const _Float16* rp = &As[t * 16 + nl][k * 32 + hi * 8];
            a.h[0] = *(const v8h*)(rp);
            a.h[1] = *(const v8h*)(rp + 16);
            acc = __builtin_amdgcn_wmma_f32_16x16x32_f16(
                false, a.v, false, B1f[k].v, (short)0, acc, false, false);
        }
#pragma unroll
        for (int r = 0; r < 8; ++r) {
            float x = acc[r] + bias1;
            float y = x / (1.0f + __expf(-x));     // silu
            Hs[t][mrow + r][w * 16 + nl] = (_Float16)y;
        }
    }
    __syncthreads();

    float* outS = out;
    float* outV = out + (size_t)NN * SDIM;

    // GEMM2 (16x128 @ 128x16 per wave) + bias + scatter-add.
#pragma unroll
    for (int t = 0; t < 4; ++t) {
        v8f acc = {};
#pragma unroll
        for (int k = 0; k < 4; ++k) {
            V16 a;
            const _Float16* rp = &Hs[t][nl][k * 32 + hi * 8];
            a.h[0] = *(const v8h*)(rp);
            a.h[1] = *(const v8h*)(rp + 16);
            acc = __builtin_amdgcn_wmma_f32_16x16x32_f16(
                false, a.v, false, B2f[k].v, (short)0, acc, false, false);
        }
        const int n = w * 16 + nl;   // raw_msg column
#pragma unroll
        for (int r = 0; r < 8; ++r) {
            float val = acc[r] + bias2;
            int   e   = t * 16 + mrow + r;
            int   dst = sDst[e];
            if (n < SDIM) atomicAdd(&outS[dst * SDIM + n], val);
            else          atomicAdd(&outV[dst * VDIM + (n - SDIM)], val);
        }
    }
}

extern "C" void kernel_launch(void* const* d_in, const int* in_sizes, int n_in,
                              void* d_out, int out_size, void* d_ws, size_t ws_size,
                              hipStream_t stream) {
    const float* hs  = (const float*)d_in[0];
    const float* hv  = (const float*)d_in[1];
    const float* pos = (const float*)d_in[2];
    const float* ori = (const float*)d_in[3];
    const float* W1  = (const float*)d_in[4];
    const float* b1  = (const float*)d_in[5];
    const float* W2  = (const float*)d_in[6];
    const float* b2  = (const float*)d_in[7];
    const int*   ei  = (const int*)d_in[8];
    float* out       = (float*)d_out;

    _Float16* W1p = (_Float16*)d_ws;                    // 7*8*32*16 halves = 57,344 B
    _Float16* W2p = W1p + 7 * 8 * 32 * 16;              // 4*8*32*16 halves = 32,768 B

    pack_w1<<<(7 * 8 * 32 * 16 + 255) / 256, 256, 0, stream>>>(W1, W1p);
    pack_w2<<<(4 * 8 * 32 * 16 + 255) / 256, 256, 0, stream>>>(W2, W2p);
    init_out<<<(NN * SDIM + 255) / 256, 256, 0, stream>>>(hs, hv, out);
    edge_kernel<<<NE / EPB, 256, 0, stream>>>(hs, hv, pos, ori, b1, b2, ei,
                                              W1p, W2p, out);
}
